// QuantLstmLayer_8710193676874
// MI455X (gfx1250) — matrix operand order repaired and checked by reading, not compiled, not run on hardware
//
#include <hip/hip_runtime.h>
#include <cstdint>
#include <cstddef>

typedef __attribute__((ext_vector_type(16))) __bf16 v16bf;
typedef __attribute__((ext_vector_type(8)))  float  v8f;

#define WMMA_BF16(A, B, C) \
  __builtin_amdgcn_wmma_f32_16x16x32_bf16(false, (A), false, (B), (short)0, (C), false, false)

constexpr int Bz = 8, T = 4096, D = 1024, H = 1024, G4 = 4096;
constexpr int M  = Bz * T;  // 32768 rows of x

// Permute K within a 32-chunk so a lane's 16 A-fragment elements are contiguous.
// A 16x32 bf16 layout (ISA 7.12.2): lane khalf=0 needs K {0..7,16..23}, khalf=1 needs {8..15,24..31}.
// Swapping 8-groups 1<->2 makes both reads a single contiguous 32B run.
__device__ __forceinline__ int permK(int k) {
  int g = (k >> 3) & 3;
  g = ((g & 1) << 1) | (g >> 1);  // 0->0, 1->2, 2->1, 3->3
  return (k & 7) | (g << 3);
}

__device__ __forceinline__ float sigmoidf_(float v) { return 1.0f / (1.0f + __expf(-v)); }
__device__ __forceinline__ float tanhf_(float v) {
  float a = fabsf(v);
  float e = __expf(-2.0f * a);
  float t = (1.0f - e) / (1.0f + e);
  return copysignf(t, v);
}

// ---------------------------------------------------------------------------
// Phase 1: x_proj[b*T+t, g] = sum_d x[b,t,d] * W_ih[g,d] + (b_ih[g]+b_hh[g])
// 128x128 tile per 256-thread block, K chunks of 32, split-bf16 (3 WMMAs).
// ---------------------------------------------------------------------------
__global__ __launch_bounds__(256) void proj_gemm(
    const float* __restrict__ x, const float* __restrict__ Wih,
    const float* __restrict__ bih, const float* __restrict__ bhh,
    float* __restrict__ xproj) {
  __shared__ __align__(32) __bf16 Ahi[128 * 32], Alo[128 * 32];
  __shared__ __align__(32) __bf16 Bhi[128 * 32], Blo[128 * 32];

  const int tid = threadIdx.x;
  const int lane = tid & 31, wave = tid >> 5;
  const int mw = wave >> 1, nw = wave & 1;           // 4x2 wave grid -> 32x64 per wave
  const int m0 = blockIdx.x * 128, n0 = blockIdx.y * 128;
  const int khalf = lane >> 4, lidx = lane & 15;

  v8f zero8 = {};
  v8f acc[2][4];
#pragma unroll
  for (int a = 0; a < 2; ++a)
#pragma unroll
    for (int b = 0; b < 4; ++b) acc[a][b] = zero8;

  for (int kb = 0; kb < D; kb += 32) {
    // Stage + convert fp32 -> bf16 hi/lo. A and B both permuted for fragment reads.
    for (int j = tid; j < 1024; j += 256) {
      int row = j >> 3, kq = (j & 7) * 4;
      const float4 av = *(const float4*)(x + (size_t)(m0 + row) * D + kb + kq);
      const float4 bv = *(const float4*)(Wih + (size_t)(n0 + row) * D + kb + kq);
      float af[4] = {av.x, av.y, av.z, av.w};
      float bf[4] = {bv.x, bv.y, bv.z, bv.w};
#pragma unroll
      for (int e = 0; e < 4; ++e) {
        int k = kq + e;
        __bf16 ah = (__bf16)af[e];
        Ahi[row * 32 + permK(k)] = ah;
        Alo[row * 32 + permK(k)] = (__bf16)(af[e] - (float)ah);
        __bf16 bh = (__bf16)bf[e];
        Bhi[row * 32 + permK(k)] = bh;
        Blo[row * 32 + permK(k)] = (__bf16)(bf[e] - (float)bh);
      }
      if (kb + 32 < D) {  // prefetch next K chunk (global_prefetch_b8)
        __builtin_prefetch(x + (size_t)(m0 + row) * D + kb + 32 + kq, 0, 1);
        __builtin_prefetch(Wih + (size_t)(n0 + row) * D + kb + 32 + kq, 0, 1);
      }
    }
    __syncthreads();

    v16bf afh[2], afl[2], bfh[4], bfl[4];
#pragma unroll
    for (int mt = 0; mt < 2; ++mt) {
      int off = (mw * 32 + mt * 16 + lidx) * 32 + khalf * 16;
      afh[mt] = *(const v16bf*)(Ahi + off);
      afl[mt] = *(const v16bf*)(Alo + off);
    }
#pragma unroll
    for (int nt = 0; nt < 4; ++nt) {
      int off = (nw * 64 + nt * 16 + lidx) * 32 + khalf * 16;
      bfh[nt] = *(const v16bf*)(Bhi + off);
      bfl[nt] = *(const v16bf*)(Blo + off);
    }
#pragma unroll
    for (int mt = 0; mt < 2; ++mt)
#pragma unroll
      for (int nt = 0; nt < 4; ++nt) {
        acc[mt][nt] = WMMA_BF16(afh[mt], bfh[nt], acc[mt][nt]);
        acc[mt][nt] = WMMA_BF16(afl[mt], bfh[nt], acc[mt][nt]);
        acc[mt][nt] = WMMA_BF16(afh[mt], bfl[nt], acc[mt][nt]);
      }
    __syncthreads();
  }

  // Epilogue: C layout (VGPR r: M = khalf*8 + r, N = lidx). Add bias, NT-store.
#pragma unroll
  for (int mt = 0; mt < 2; ++mt) {
    int gmBase = m0 + mw * 32 + mt * 16 + khalf * 8;
#pragma unroll
    for (int nt = 0; nt < 4; ++nt) {
      int gn = n0 + nw * 64 + nt * 16 + lidx;
      float bias = bih[gn] + bhh[gn];
#pragma unroll
      for (int r = 0; r < 8; ++r)
        __builtin_nontemporal_store(acc[mt][nt][r] + bias,
                                    xproj + (size_t)(gmBase + r) * G4 + gn);
    }
  }
}

// ---------------------------------------------------------------------------
// Phase 2: sequential scan. 64 WGs x 128 threads; WG wg owns units [wg*16, wg*16+16);
// wave g (0..3) owns gate g's 16 columns. W_hh slice persists in LDS (bf16 hi+lo).
// h ping-pongs in global fp32; one device-scope barrier per timestep.
// ---------------------------------------------------------------------------
__global__ __launch_bounds__(128) void lstm_scan(
    const float* __restrict__ xproj, const float* __restrict__ Whh,
    const float* __restrict__ h0, const float* __restrict__ c0,
    const int* __restrict__ lens, float* __restrict__ out,
    float* __restrict__ hbuf0, float* __restrict__ hbuf1,
    unsigned* __restrict__ bar_cnt, unsigned* __restrict__ bar_gen) {
  extern __shared__ __align__(32) char smem[];
  __bf16* whh_hi = (__bf16*)smem;              // [4][16][1024]
  __bf16* whh_lo = whh_hi + 4 * 16 * 1024;     // [4][16][1024]
  __bf16* hhi    = whh_lo + 4 * 16 * 1024;     // [8][1024], permuted A layout
  __bf16* hlo    = hhi + 8 * 1024;
  float*  gst    = (float*)(hlo + 8 * 1024);   // [4 gates][8 batch][16 units]

  const int tid = threadIdx.x, lane = tid & 31, wave = tid >> 5;
  const int u0 = blockIdx.x * 16;
  const int khalf = lane >> 4, lidx = lane & 15;

  // Prologue: convert this WG's W_hh slice to bf16 hi/lo in LDS (once, reused 4096 steps).
  for (int j4 = lane; j4 < (16 * 1024) / 4; j4 += 32) {
    int n = j4 >> 8, kq = (j4 & 255) * 4;
    const float4 wv = *(const float4*)(Whh + (size_t)((wave << 10) + u0 + n) * H + kq);
    float wf[4] = {wv.x, wv.y, wv.z, wv.w};
#pragma unroll
    for (int e = 0; e < 4; ++e) {
      int k = kq + e;
      __bf16 wh = (__bf16)wf[e];
      whh_hi[(wave << 14) + (n << 10) + (k & ~31) + permK(k & 31)] = wh;
      whh_lo[(wave << 14) + (n << 10) + (k & ~31) + permK(k & 31)] = (__bf16)(wf[e] - (float)wh);
    }
  }

  const int b = tid >> 4, u = tid & 15;  // 128 threads = 8 batches x 16 units
  const int myLen = lens[b];
  float creg = c0[b * H + u0 + u];
  float hreg = h0[b * H + u0 + u];
  __syncthreads();

  for (int t = 0; t < T; ++t) {
    const float* hsrc = (t & 1) ? hbuf1 : hbuf0;
    float* hdst       = (t & 1) ? hbuf0 : hbuf1;

    // Stage h (8x1024 fp32) into LDS as split-bf16 in permuted A-fragment layout.
    for (int j4 = tid; j4 < (8 * 1024) / 4; j4 += 128) {
      int m = j4 >> 8, kq = (j4 & 255) * 4;
      const float4 hv = *(const float4*)(hsrc + m * H + kq);
      float hf[4] = {hv.x, hv.y, hv.z, hv.w};
#pragma unroll
      for (int e = 0; e < 4; ++e) {
        int k = kq + e;
        int slot = (m << 10) + (k & ~31) + permK(k & 31);
        __bf16 hh = (__bf16)hf[e];
        hhi[slot] = hh;
        hlo[slot] = (__bf16)(hf[e] - (float)hh);
      }
    }
    __syncthreads();

    // Gate pre-activations: wave's 16 columns, K = 1024 in 32-chunks, 3 WMMAs each.
    v8f acc = {};
    const __bf16* wbh = whh_hi + (wave << 14) + (lidx << 10);
    const __bf16* wbl = whh_lo + (wave << 14) + (lidx << 10);
    const __bf16* abh = hhi + ((lidx & 7) << 10);  // clamp row; rows >=8 zeroed below
    const __bf16* abl = hlo + ((lidx & 7) << 10);
    const bool padrow = (lidx >= 8);
#pragma unroll 4
    for (int kb = 0; kb < H; kb += 32) {
      v16bf ah = *(const v16bf*)(abh + kb + khalf * 16);
      v16bf al = *(const v16bf*)(abl + kb + khalf * 16);
      if (padrow) { v16bf z = {}; ah = z; al = z; }
      v16bf bh = *(const v16bf*)(wbh + kb + khalf * 16);
      v16bf bl = *(const v16bf*)(wbl + kb + khalf * 16);
      acc = WMMA_BF16(ah, bh, acc);
      acc = WMMA_BF16(al, bh, acc);
      acc = WMMA_BF16(ah, bl, acc);
    }
    if (khalf == 0) {  // lanes 0..15 hold M = r (batches 0..7) in VGPR r
#pragma unroll
      for (int r = 0; r < 8; ++r) gst[(wave << 7) + (r << 4) + lidx] = acc[r];
    }
    __syncthreads();

    // Elementwise LSTM cell for (b, u). Gate order i,f,g,o.
    {
      const size_t xo = ((size_t)(b * T + t)) * G4 + u0 + u;
      float gi = gst[0 * 128 + b * 16 + u] + __builtin_nontemporal_load(xproj + xo + 0 * H);
      float gf = gst[1 * 128 + b * 16 + u] + __builtin_nontemporal_load(xproj + xo + 1 * H);
      float gg = gst[2 * 128 + b * 16 + u] + __builtin_nontemporal_load(xproj + xo + 2 * H);
      float go = gst[3 * 128 + b * 16 + u] + __builtin_nontemporal_load(xproj + xo + 3 * H);
      float cn = sigmoidf_(gf) * creg + sigmoidf_(gi) * tanhf_(gg);
      float hn = sigmoidf_(go) * tanhf_(cn);
      bool active = (t < myLen);
      if (active) { creg = cn; hreg = hn; }
      out[((size_t)(b * T + t)) * H + u0 + u] = active ? hn : 0.0f;
      hdst[b * H + u0 + u] = hreg;  // always publish (carry h forward when inactive)
    }

    // Device-scope grid barrier: all h writes visible before any WG stages step t+1.
    __threadfence();
    __syncthreads();
    if (tid == 0) {
      unsigned g = __hip_atomic_load(bar_gen, __ATOMIC_RELAXED, __HIP_MEMORY_SCOPE_AGENT);
      unsigned a = __hip_atomic_fetch_add(bar_cnt, 1u, __ATOMIC_ACQ_REL, __HIP_MEMORY_SCOPE_AGENT);
      if (a == (unsigned)gridDim.x - 1u) {
        __hip_atomic_store(bar_cnt, 0u, __ATOMIC_RELAXED, __HIP_MEMORY_SCOPE_AGENT);
        __hip_atomic_store(bar_gen, g + 1u, __ATOMIC_RELEASE, __HIP_MEMORY_SCOPE_AGENT);
      } else {
        while (__hip_atomic_load(bar_gen, __ATOMIC_ACQUIRE, __HIP_MEMORY_SCOPE_AGENT) == g)
          __builtin_amdgcn_s_sleep(1);
      }
    }
    __syncthreads();
    __threadfence();
  }

  // Final states: h_n then c_n, appended after output[B,T,H].
  out[(size_t)Bz * T * H + b * H + u0 + u] = hreg;
  out[(size_t)Bz * T * H + (size_t)Bz * H + b * H + u0 + u] = creg;
}

// ---------------------------------------------------------------------------
extern "C" void kernel_launch(void* const* d_in, const int* in_sizes, int n_in,
                              void* d_out, int out_size, void* d_ws, size_t ws_size,
                              hipStream_t stream) {
  (void)in_sizes; (void)n_in; (void)out_size; (void)ws_size;
  const float* x    = (const float*)d_in[0];
  const float* h0   = (const float*)d_in[1];
  const float* c0   = (const float*)d_in[2];
  const float* Wih  = (const float*)d_in[3];
  const float* Whh  = (const float*)d_in[4];
  const float* bih  = (const float*)d_in[5];
  const float* bhh  = (const float*)d_in[6];
  const int*   lens = (const int*)d_in[7];
  float* out = (float*)d_out;

  char* ws = (char*)d_ws;
  float* xproj = (float*)ws;  // [M][4H] fp32 = 512 MiB
  size_t off = (size_t)M * G4 * sizeof(float);
  float* hbuf0 = (float*)(ws + off);  off += (size_t)Bz * H * sizeof(float);
  float* hbuf1 = (float*)(ws + off);  off += (size_t)Bz * H * sizeof(float);
  unsigned* bar = (unsigned*)(ws + off);

  hipMemcpyAsync(hbuf0, h0, (size_t)Bz * H * sizeof(float), hipMemcpyDeviceToDevice, stream);
  hipMemsetAsync(bar, 0, 2 * sizeof(unsigned), stream);

  dim3 gp(M / 128, G4 / 128);  // 256 x 32 blocks
  proj_gemm<<<gp, 256, 0, stream>>>(x, Wih, bih, bhh, xproj);

  constexpr int SMEM =
      (4 * 16 * 1024 + 4 * 16 * 1024 + 8 * 1024 + 8 * 1024) * 2 + 4 * 8 * 16 * 4;  // 296960 B
  hipFuncSetAttribute((const void*)lstm_scan, hipFuncAttributeMaxDynamicSharedMemorySize, SMEM);
  lstm_scan<<<64, 128, SMEM, stream>>>(xproj, Whh, h0, c0, lens, out, hbuf0, hbuf1,
                                       bar, bar + 1);
}
